// ConformerMHSAV3_36464272343407
// MI455X (gfx1250) — matrix operand order repaired and checked by reading, not compiled, not run on hardware
//
#include <hip/hip_runtime.h>
#include <math.h>

// ---------------------------------------------------------------------------
// ConformerMHSA (dual-mask) for MI455X / gfx1250.
// B=4 T=1024 D=1024 H=16 HD=64.  ~70 GFLOP vs ~100MB traffic -> matrix-core
// bound.  All GEMM-like math runs on v_wmma_f32_16x16x32_bf16 (bf16 in, f32
// accumulate); LN / softmax / epilogues stay fp32.  LDS tile fills use the
// gfx1250 async global->LDS path (ASYNCcnt) where no transpose is needed.
// ---------------------------------------------------------------------------

typedef __attribute__((ext_vector_type(16))) __bf16 v16bf;
typedef __attribute__((ext_vector_type(8)))  float  v8f;

#define CB  4
#define CT  1024
#define CD  1024
#define CH  16
#define CHD 64
#define C3D 3072
#define CBT (CB * CT)          // 4096 rows
#define EPSV 1e-05f

// ---------------------------------------------------------------------------
// Async global -> LDS copy (GLOBAL_LOAD_ASYNC_TO_LDS_B128, ISA 10/15.18.3).
// Per active lane: LDS[ldsoff] = MEM[gaddr] for 16 bytes.  The low 32 bits of
// a generic shared-aperture pointer are the LDS byte offset (aperture check
// truncates ADDR[31:0]), so casting a __shared__ pointer works as VDST.
// Tracked with ASYNCcnt -> s_wait_asynccnt 0 before the consuming barrier.
// ---------------------------------------------------------------------------
__device__ __forceinline__ void async_g2l_b128(void* ldsptr, const void* gptr) {
  unsigned int ldsoff = (unsigned int)(unsigned long long)ldsptr;
  unsigned long long gaddr = (unsigned long long)gptr;
  asm volatile("global_load_async_to_lds_b128 %0, %1, off"
               :: "v"(ldsoff), "v"(gaddr) : "memory");
}
__device__ __forceinline__ void wait_async0() {
  asm volatile("s_wait_asynccnt 0" ::: "memory");
}

// ---------------------------------------------------------------------------
// WMMA helpers.
// C/D layout (16x16 f32, 8 VGPRs): element v of lane l is (row = v + 8*(l>>4),
// col = l&15).
// A layout (16x32 bf16): lane<16 -> row=lane, K 0..7 & 16..23; lane>=16 ->
// row=lane-16, K 8..15 & 24..31.   (ISA 7.12.2 table)
// B layout (32x16 bf16): lane<16 -> col=lane, K 0..7 & 16..23; lane>=16 ->
// col=lane-16, K 8..15 & 24..31.   (same K order as A)
// ---------------------------------------------------------------------------
__device__ __forceinline__ v8f wmma_bf16(v16bf a, v16bf b, v8f c) {
  return __builtin_amdgcn_wmma_f32_16x16x32_bf16(
      /*neg_a=*/false, a, /*neg_b=*/false, b,
      /*c_mod=*/(short)0, c, /*reuse_a=*/false, /*reuse_b=*/false);
}

// src: row-major [16 rows][>=32 cols] bf16 tile in LDS; ld in elements,
// (ld*2) % 16 must be 0.
__device__ __forceinline__ v16bf load_fragA(const __bf16* base, int ld) {
  const int lane = threadIdx.x & 31;
  const int r    = lane & 15;
  const int kh   = (lane >> 4) & 1;            // 0: K0-7 & 16-23, 1: K8-15 & 24-31
  const __bf16* p = base + r * ld + kh * 8;
  union { v16bf v; uint4 u[2]; } f;
  f.u[0] = *(const uint4*)(p);
  f.u[1] = *(const uint4*)(p + 16);
  return f.v;
}

// src: row-major [16 n-rows][>=32 k-cols] bf16 tile in LDS (row = N, col = K).
__device__ __forceinline__ v16bf load_fragB(const __bf16* base, int ld) {
  const int lane = threadIdx.x & 31;
  const __bf16* p = base + (lane & 15) * ld + ((lane >> 4) & 1) * 8;
  union { v16bf v; uint4 u[2]; } f;
  f.u[0] = *(const uint4*)(p);
  f.u[1] = *(const uint4*)(p + 16);
  return f.v;
}

__device__ __forceinline__ v8f v8f_zero() {
  v8f z = {0.f, 0.f, 0.f, 0.f, 0.f, 0.f, 0.f, 0.f};
  return z;
}

// ---------------------------------------------------------------------------
// fp32 -> bf16 cast (weights)
// ---------------------------------------------------------------------------
__global__ __launch_bounds__(256) void cast_bf16_kernel(
    const float* __restrict__ in, __bf16* __restrict__ out, int n) {
  for (int i = blockIdx.x * blockDim.x + threadIdx.x; i < n;
       i += gridDim.x * blockDim.x)
    out[i] = (__bf16)in[i];
}

// ---------------------------------------------------------------------------
// LayerNorm over D=1024, one block (8 waves) per row, emits bf16.
// ---------------------------------------------------------------------------
__global__ __launch_bounds__(256) void ln_kernel(
    const float* __restrict__ x, const float* __restrict__ gamma,
    const float* __restrict__ beta, __bf16* __restrict__ y) {
  const int row = blockIdx.x;
  const int tid = threadIdx.x;
  const float* xr = x + (size_t)row * CD;

  float vals[4];
  float s = 0.f, sq = 0.f;
#pragma unroll
  for (int i = 0; i < 4; ++i) {
    float v = xr[tid + i * 256];
    vals[i] = v;
    s += v;
    sq += v * v;
  }
#pragma unroll
  for (int off = 16; off >= 1; off >>= 1) {
    s  += __shfl_xor(s,  off, 32);
    sq += __shfl_xor(sq, off, 32);
  }
  __shared__ float ps[8], pq[8];
  const int wid = tid >> 5, lane = tid & 31;
  if (lane == 0) { ps[wid] = s; pq[wid] = sq; }
  __syncthreads();
  s = 0.f; sq = 0.f;
#pragma unroll
  for (int w = 0; w < 8; ++w) { s += ps[w]; sq += pq[w]; }
  const float mu   = s * (1.0f / CD);
  const float var  = sq * (1.0f / CD) - mu * mu;
  const float rstd = rsqrtf(var + EPSV);

  __bf16* yr = y + (size_t)row * CD;
#pragma unroll
  for (int i = 0; i < 4; ++i) {
    int c = tid + i * 256;
    yr[c] = (__bf16)((vals[i] - mu) * rstd * gamma[c] + beta[c]);
  }
}

// ---------------------------------------------------------------------------
// Generic bf16 WMMA GEMM:  C[M][N] = A[M][K] * W[N][K]^T + bias[N]
// Block tile 128x128, 8 waves (4 m x 2 n), wave tile 32x64, BK=32.
// Tile fills ride the async global->LDS pipe.
// ---------------------------------------------------------------------------
#define GBM 128
#define GBN 128
#define GBK 32
#define GLD 48   // LDS row stride (elems); 96B per row, 16B-aligned chunks

template <typename OutT>
__global__ __launch_bounds__(256) void gemm_bf16_kernel(
    const __bf16* __restrict__ A, const __bf16* __restrict__ W,
    const float* __restrict__ bias, OutT* __restrict__ C,
    int M, int N, int K) {
  __shared__ __align__(16) __bf16 sA[GBM * GLD];
  __shared__ __align__(16) __bf16 sB[GBN * GLD];

  const int tid  = threadIdx.x;
  const int wid  = tid >> 5;
  const int lane = tid & 31;
  const int wm   = wid >> 1;    // 0..3
  const int wn   = wid & 1;     // 0..1
  const int m0   = blockIdx.y * GBM;
  const int n0   = blockIdx.x * GBN;

  v8f acc[2][4];
#pragma unroll
  for (int i = 0; i < 2; ++i)
#pragma unroll
    for (int j = 0; j < 4; ++j) acc[i][j] = v8f_zero();

  for (int k0 = 0; k0 < K; k0 += GBK) {
    __syncthreads();
    // A tile: 128 rows x 32 cols = 512 x 16B async chunks
#pragma unroll
    for (int i = tid; i < GBM * 4; i += 256) {
      int r = i >> 2, ch = i & 3;
      async_g2l_b128(&sA[r * GLD + ch * 8],
                     A + (size_t)(m0 + r) * K + k0 + ch * 8);
    }
    // W tile: 128 n-rows x 32 k-cols
#pragma unroll
    for (int i = tid; i < GBN * 4; i += 256) {
      int r = i >> 2, ch = i & 3;
      async_g2l_b128(&sB[r * GLD + ch * 8],
                     W + (size_t)(n0 + r) * K + k0 + ch * 8);
    }
    wait_async0();
    __syncthreads();

    v16bf af[2], bf[4];
#pragma unroll
    for (int i = 0; i < 2; ++i)
      af[i] = load_fragA(&sA[(wm * 32 + i * 16) * GLD], GLD);
#pragma unroll
    for (int j = 0; j < 4; ++j)
      bf[j] = load_fragB(&sB[(wn * 64 + j * 16) * GLD], GLD);
#pragma unroll
    for (int i = 0; i < 2; ++i)
#pragma unroll
      for (int j = 0; j < 4; ++j) acc[i][j] = wmma_bf16(af[i], bf[j], acc[i][j]);
  }

  const int half = lane >> 4, col = lane & 15;
#pragma unroll
  for (int i = 0; i < 2; ++i)
#pragma unroll
    for (int j = 0; j < 4; ++j) {
      const int gc = n0 + wn * 64 + j * 16 + col;
      const float bv = bias[gc];
#pragma unroll
      for (int v = 0; v < 8; ++v) {
        const int gr = m0 + wm * 32 + i * 16 + v + 8 * half;
        C[(size_t)gr * N + gc] = (OutT)(acc[i][j][v] + bv);
      }
    }
}

// ---------------------------------------------------------------------------
// Dual-mask flash attention.
// qkv: bf16 [B*T][3072] (q | k | v, head-interleaved by 64).
// O:   bf16 [2][B*T][D], pass 0 = causal+pad, pass 1 = pad-only.
// Block = 4 waves; each wave owns a 16-row query tile; K/V tiles shared in LDS.
// ---------------------------------------------------------------------------
#define FLDK 72   // K tile [32 keys][64 hd], 144B rows
#define FLDV 48   // V^T tile [64 hd][32 keys], 96B rows
#define FLDQ 72   // Q stage [64 q][64 hd]
#define FLDP 48   // P scratch [16 q][32 keys]

__global__ __launch_bounds__(128) void flash_dual_kernel(
    const __bf16* __restrict__ qkv, const unsigned char* __restrict__ seqmask,
    __bf16* __restrict__ O) {
  __shared__ __align__(16) __bf16 sQ[64 * FLDQ];
  __shared__ __align__(16) __bf16 sK[32 * FLDK];
  __shared__ __align__(16) __bf16 sV[64 * FLDV];         // transposed [hd][key]
  __shared__ __align__(16) __bf16 sP[4][16 * FLDP];      // per-wave scratch

  const int b  = blockIdx.z;
  const int h  = blockIdx.y;
  const int qb = blockIdx.x * 64;
  const int tid = threadIdx.x, wid = tid >> 5, lane = tid & 31;
  const int half = lane >> 4, l16 = lane & 15;
  const float NEG_INF = -__builtin_inff();

  const __bf16* Qg = qkv + (size_t)b * CT * C3D + (size_t)h * CHD;
  const __bf16* Kg = Qg + CD;
  const __bf16* Vg = Qg + 2 * CD;
  const unsigned char* sm = seqmask + b * CT;

  // Stage this block's 64 query rows once (async, 512 x 16B chunks).
#pragma unroll
  for (int i = tid; i < 64 * 8; i += 128) {
    int r = i >> 3, c = i & 7;
    async_g2l_b128(&sQ[r * FLDQ + c * 8],
                   Qg + (size_t)(qb + r) * C3D + c * 8);
  }
  wait_async0();
  __syncthreads();
  v16bf qf[2];
  qf[0] = load_fragA(&sQ[(wid * 16) * FLDQ + 0], FLDQ);
  qf[1] = load_fragA(&sQ[(wid * 16) * FLDQ + 32], FLDQ);

  v8f   o[2][4];
  float m[2][8], l[2][8];
#pragma unroll
  for (int p = 0; p < 2; ++p) {
#pragma unroll
    for (int j = 0; j < 4; ++j) o[p][j] = v8f_zero();
#pragma unroll
    for (int v = 0; v < 8; ++v) { m[p][v] = NEG_INF; l[p][v] = 0.f; }
  }

  for (int kt = 0; kt < CT; kt += 32) {
    __syncthreads();   // previous tiles fully consumed
    // K tile: 32 rows x 64 hd (async, straight copy)
#pragma unroll
    for (int i = tid; i < 32 * 8; i += 128) {
      int r = i >> 3, c = i & 7;
      async_g2l_b128(&sK[r * FLDK + c * 8],
                     Kg + (size_t)(kt + r) * C3D + c * 8);
    }
    // V tile transposed: [hd][key] (manual path: needs transpose in flight)
    for (int i = tid; i < 32 * 4; i += 128) {
      int r = i >> 2, c = i & 3;   // key r, hd chunk c*16
      union { uint4 u[2]; __bf16 e[16]; } dv;
      const __bf16* vp = Vg + (size_t)(kt + r) * C3D + c * 16;
      dv.u[0] = *(const uint4*)(vp);
      dv.u[1] = *(const uint4*)(vp + 8);
#pragma unroll
      for (int e = 0; e < 16; ++e) sV[(c * 16 + e) * FLDV + r] = dv.e[e];
    }
    wait_async0();
    __syncthreads();

    // S = Q K^T for 32 keys -> two 16x16 f32 tiles.
    v8f s[2];
#pragma unroll
    for (int j = 0; j < 2; ++j) {
      v16bf b0 = load_fragB(&sK[(j * 16) * FLDK + 0], FLDK);
      v16bf b1 = load_fragB(&sK[(j * 16) * FLDK + 32], FLDK);
      v8f c = v8f_zero();
      c = wmma_bf16(qf[0], b0, c);
      c = wmma_bf16(qf[1], b1, c);
      s[j] = c;
    }

    const int  kg0  = kt + l16;
    const int  kg1  = kt + 16 + l16;
    const bool pad0 = sm[kg0] != 0;
    const bool pad1 = sm[kg1] != 0;

    for (int pass = 0; pass < 2; ++pass) {
      float pe[2][8];
#pragma unroll
      for (int v = 0; v < 8; ++v) {
        const int qg = qb + wid * 16 + v + 8 * half;
        float s0 = s[0][v] * 0.125f;      // 1/sqrt(HD)
        float s1 = s[1][v] * 0.125f;
        const bool ok0 = pad0 && (pass == 1 || kg0 <= qg);
        const bool ok1 = pad1 && (pass == 1 || kg1 <= qg);
        s0 = ok0 ? s0 : NEG_INF;
        s1 = ok1 ? s1 : NEG_INF;
        float mx = fmaxf(s0, s1);
#pragma unroll
        for (int off = 8; off >= 1; off >>= 1)
          mx = fmaxf(mx, __shfl_xor(mx, off, 16));
        const float mnew = fmaxf(m[pass][v], mx);
        float corr, e0, e1;
        if (mnew == NEG_INF) {            // fully-masked-so-far row
          corr = 0.f; e0 = 0.f; e1 = 0.f;
        } else {
          corr = __expf(m[pass][v] - mnew);
          e0 = ok0 ? __expf(s0 - mnew) : 0.f;
          e1 = ok1 ? __expf(s1 - mnew) : 0.f;
        }
        float rs = e0 + e1;
#pragma unroll
        for (int off = 8; off >= 1; off >>= 1)
          rs += __shfl_xor(rs, off, 16);
        l[pass][v] = l[pass][v] * corr + rs;
        m[pass][v] = mnew;
        pe[0][v] = e0;
        pe[1][v] = e1;
#pragma unroll
        for (int jh = 0; jh < 4; ++jh) o[pass][jh][v] *= corr;
      }
      // C-layout -> A-layout through per-wave LDS scratch (DS is in-order per
      // wave; the explicit wait + scheduling barrier keep the compiler honest).
#pragma unroll
      for (int j = 0; j < 2; ++j)
#pragma unroll
        for (int v = 0; v < 8; ++v)
          sP[wid][(v + 8 * half) * FLDP + j * 16 + l16] = (__bf16)pe[j][v];
      asm volatile("s_wait_dscnt 0" ::: "memory");
      __builtin_amdgcn_wave_barrier();
      v16bf pf = load_fragA(&sP[wid][0], FLDP);
      __builtin_amdgcn_wave_barrier();
#pragma unroll
      for (int jh = 0; jh < 4; ++jh) {
        v16bf vf = load_fragB(&sV[(jh * 16) * FLDV], FLDV);
        o[pass][jh] = wmma_bf16(pf, vf, o[pass][jh]);
      }
    }
  }

  // Normalize and write bf16 O for the output projection GEMM.
#pragma unroll
  for (int pass = 0; pass < 2; ++pass)
#pragma unroll
    for (int v = 0; v < 8; ++v) {
      const float inv = l[pass][v] > 0.f ? 1.0f / l[pass][v] : 0.f;
      const int qg = qb + wid * 16 + v + 8 * half;
      const size_t base =
          ((size_t)pass * CBT + (size_t)b * CT + qg) * CD + (size_t)h * CHD;
#pragma unroll
      for (int jh = 0; jh < 4; ++jh)
        O[base + jh * 16 + l16] = (__bf16)(o[pass][jh][v] * inv);
    }
}

// ---------------------------------------------------------------------------
// Host-side orchestration.
// ---------------------------------------------------------------------------
extern "C" void kernel_launch(void* const* d_in, const int* in_sizes, int n_in,
                              void* d_out, int out_size, void* d_ws,
                              size_t ws_size, hipStream_t stream) {
  (void)in_sizes; (void)n_in; (void)out_size; (void)ws_size;

  const float*         x     = (const float*)d_in[0];          // (B,T,D)
  const float*         gamma = (const float*)d_in[1];          // (D,)
  const float*         beta  = (const float*)d_in[2];          // (D,)
  const float*         wqkv  = (const float*)d_in[3];          // (3D,D)
  const float*         bqkv  = (const float*)d_in[4];          // (3D,)
  const float*         wout  = (const float*)d_in[5];          // (D,D)
  const float*         bout  = (const float*)d_in[6];          // (D,)
  const unsigned char* smask = (const unsigned char*)d_in[7];  // (B,T) bool
  // d_in[8] = attn_mask (causal tril) — computed arithmetically instead.
  float* out = (float*)d_out;                                  // (2,B,T,D)

  char* ws = (char*)d_ws;
  size_t off = 0;
  __bf16* ws_x    = (__bf16*)(ws + off); off += (size_t)CBT * CD * 2;       // 8 MB
  __bf16* ws_wqkv = (__bf16*)(ws + off); off += (size_t)C3D * CD * 2;       // 6 MB
  __bf16* ws_wout = (__bf16*)(ws + off); off += (size_t)CD * CD * 2;        // 2 MB
  __bf16* ws_qkv  = (__bf16*)(ws + off); off += (size_t)CBT * C3D * 2;      // 24 MB
  __bf16* ws_o    = (__bf16*)(ws + off);                                    // 16 MB

  // 1) weights -> bf16
  cast_bf16_kernel<<<2048, 256, 0, stream>>>(wqkv, ws_wqkv, C3D * CD);
  cast_bf16_kernel<<<1024, 256, 0, stream>>>(wout, ws_wout, CD * CD);

  // 2) LayerNorm -> bf16 X
  ln_kernel<<<CBT, 256, 0, stream>>>(x, gamma, beta, ws_x);

  // 3) QKV = X @ Wqkv^T + b   (4096 x 3072 x 1024), bf16 out
  {
    dim3 grid(C3D / GBN, CBT / GBM);
    gemm_bf16_kernel<__bf16><<<grid, 256, 0, stream>>>(
        ws_x, ws_wqkv, bqkv, ws_qkv, CBT, C3D, CD);
  }

  // 4) dual-mask flash attention -> bf16 O [2][B*T][D]
  {
    dim3 grid(CT / 64, CH, CB);
    flash_dual_kernel<<<grid, 128, 0, stream>>>(ws_qkv, smask, ws_o);
  }

  // 5) out = O @ Wout^T + b   (8192 x 1024 x 1024), f32 out, both passes
  {
    dim3 grid(CD / GBN, (2 * CBT) / GBM);
    gemm_bf16_kernel<float><<<grid, 256, 0, stream>>>(
        ws_o, ws_wout, bout, out, 2 * CBT, CD, CD);
  }
}
